// Attention_54254026883778
// MI455X (gfx1250) — matrix operand order repaired and checked by reading, not compiled, not run on hardware
//
#include <hip/hip_runtime.h>

typedef __bf16 bf16;
typedef __attribute__((ext_vector_type(8)))  bf16  v8bf;
typedef __attribute__((ext_vector_type(16))) bf16  v16bf;
typedef __attribute__((ext_vector_type(8)))  float v8f;

#define TILE_M 128
#define TILE_N 128
#define TILE_K 32

// Stage one 128x32 tile (rows row0..row0+127, cols k0..k0+31) into LDS as bf16.
// 256 threads: thread t handles row t>>1, 16-element half (t&1).
template<bool SRC_F32>
__device__ __forceinline__ void stage_tile(bf16 dst[][TILE_K], const void* __restrict__ src,
                                           long ld, long row0, int k0, int tid) {
  const int row = tid >> 1;
  const int kc  = (tid & 1) * 16;
  if constexpr (SRC_F32) {
    const float* g = (const float*)src + (row0 + row) * ld + k0 + kc;
    float4 f0 = *(const float4*)(g + 0);
    float4 f1 = *(const float4*)(g + 4);
    float4 f2 = *(const float4*)(g + 8);
    float4 f3 = *(const float4*)(g + 12);
    v8bf lo, hi;
    lo[0]=(bf16)f0.x; lo[1]=(bf16)f0.y; lo[2]=(bf16)f0.z; lo[3]=(bf16)f0.w;
    lo[4]=(bf16)f1.x; lo[5]=(bf16)f1.y; lo[6]=(bf16)f1.z; lo[7]=(bf16)f1.w;
    hi[0]=(bf16)f2.x; hi[1]=(bf16)f2.y; hi[2]=(bf16)f2.z; hi[3]=(bf16)f2.w;
    hi[4]=(bf16)f3.x; hi[5]=(bf16)f3.y; hi[6]=(bf16)f3.z; hi[7]=(bf16)f3.w;
    *(v8bf*)&dst[row][kc]     = lo;
    *(v8bf*)&dst[row][kc + 8] = hi;
  } else {
    const bf16* g = (const bf16*)src + (row0 + row) * ld + k0 + kc;
    *(v8bf*)&dst[row][kc]     = *(const v8bf*)(g);
    *(v8bf*)&dst[row][kc + 8] = *(const v8bf*)(g + 8);
  }
}

// C[m][n] = scale * sum_k A[m][k] * B[n][k]  (+ bias)
// A: [M x K] (row major), B: [N x K] (row major, i.e. the transposed operand),
// biasMode: 0 = none, 1 = bias[n], 2 = bias[m].
template<bool A_F32, bool B_F32, bool OUT_F32>
__global__ __launch_bounds__(256)
void gemm_wmma_kernel(const void* __restrict__ A, long lda, long aBatch,
                      const void* __restrict__ B, long ldb, long bBatch,
                      void* __restrict__ C, long ldc, long cBatch,
                      const float* __restrict__ bias, int biasMode,
                      float scale, int K) {
  __shared__ __align__(64) bf16 As[TILE_M][TILE_K];
  __shared__ __align__(64) bf16 Bs[TILE_N][TILE_K];

  const int tid  = threadIdx.x;
  const int lane = tid & 31;
  const int wid  = tid >> 5;          // 8 wave32 per block
  const int wm   = wid >> 1;          // 0..3  -> 32-row slab
  const int wn   = wid & 1;           // 0..1  -> 64-col slab
  const int lm   = lane & 15;
  const int lg   = lane >> 4;

  const long m0 = (long)blockIdx.y * TILE_M;
  const long n0 = (long)blockIdx.x * TILE_N;
  const long bz = (long)blockIdx.z;

  const void* Ap;
  const void* Bp;
  void* Cp;
  if constexpr (A_F32) Ap = (const float*)A + bz * aBatch; else Ap = (const bf16*)A + bz * aBatch;
  if constexpr (B_F32) Bp = (const float*)B + bz * bBatch; else Bp = (const bf16*)B + bz * bBatch;
  if constexpr (OUT_F32) Cp = (float*)C + bz * cBatch; else Cp = (bf16*)C + bz * cBatch;

  v8f acc[2][4];
#pragma unroll
  for (int i = 0; i < 2; ++i)
#pragma unroll
    for (int j = 0; j < 4; ++j)
#pragma unroll
      for (int r = 0; r < 8; ++r) acc[i][j][r] = 0.0f;

  for (int k0 = 0; k0 < K; k0 += TILE_K) {
    stage_tile<A_F32>(As, Ap, lda, m0, k0, tid);
    stage_tile<B_F32>(Bs, Bp, ldb, n0, k0, tid);

    // prefetch next K tile rows into cache (global_prefetch_b8)
    if (k0 + TILE_K < K) {
      const int row = tid >> 1;
      if constexpr (A_F32) __builtin_prefetch((const float*)Ap + (m0 + row) * lda + k0 + TILE_K, 0, 1);
      else                 __builtin_prefetch((const bf16*)Ap + (m0 + row) * lda + k0 + TILE_K, 0, 1);
      if constexpr (B_F32) __builtin_prefetch((const float*)Bp + (n0 + row) * ldb + k0 + TILE_K, 0, 1);
      else                 __builtin_prefetch((const bf16*)Bp + (n0 + row) * ldb + k0 + TILE_K, 0, 1);
    }
    __syncthreads();

    // A fragments: lane lm holds row (..+lm); elements 0..15 = K[lg*16..lg*16+15] (contiguous per lane)
    v16bf afrag[2];
#pragma unroll
    for (int i = 0; i < 2; ++i) {
      const bf16* p = &As[wm * 32 + i * 16 + lm][0];
      v8bf lo = *(const v8bf*)(p + lg * 16);
      v8bf hi = *(const v8bf*)(p + lg * 16 + 8);
      afrag[i] = __builtin_shufflevector(lo, hi, 0,1,2,3,4,5,6,7,8,9,10,11,12,13,14,15);
    }
    // B fragments: lane lm holds column n0+..+lm; lanes 16-31 hold K 16..31 (contiguous per lane)
    v16bf bfrag[4];
#pragma unroll
    for (int j = 0; j < 4; ++j) {
      const bf16* p = &Bs[wn * 64 + j * 16 + lm][lg * 16];
      bfrag[j] = *(const v16bf*)p;
    }
#pragma unroll
    for (int i = 0; i < 2; ++i)
#pragma unroll
      for (int j = 0; j < 4; ++j)
        acc[i][j] = __builtin_amdgcn_wmma_f32_16x16x32_bf16(
            false, afrag[i], false, bfrag[j], (short)0, acc[i][j], false, false);
    __syncthreads();
  }

  // Epilogue: lane lm = column, VGPR r -> row r + 8*lg within 16-row fragment
#pragma unroll
  for (int i = 0; i < 2; ++i) {
    const long mbase = m0 + wm * 32 + i * 16 + lg * 8;
#pragma unroll
    for (int j = 0; j < 4; ++j) {
      const long n = n0 + wn * 64 + j * 16 + lm;
      const float bn = (biasMode == 1) ? bias[n] : 0.0f;
#pragma unroll
      for (int r = 0; r < 8; ++r) {
        const long m = mbase + r;
        float v = acc[i][j][r] * scale + bn;
        if (biasMode == 2) v += bias[m];
        if constexpr (OUT_F32) ((float*)Cp)[m * ldc + n] = v;
        else                   ((bf16*)Cp)[m * ldc + n] = (bf16)v;
      }
    }
  }
}

// One 256-thread block per score row: softmax over 2048 cols, add intensity, emit bf16.
__global__ __launch_bounds__(256)
void softmax_intensity_kernel(const float* __restrict__ scores,
                              const float* __restrict__ intensity,
                              bf16* __restrict__ attn) {
  const int S = 2048;
  const int tid = threadIdx.x;
  const long base = ((long)blockIdx.y * S + blockIdx.x) * S;
  const float* x  = scores + base;
  const float* it = intensity + base;
  bf16* out = attn + base;

  float vals[8];
#pragma unroll
  for (int i = 0; i < 8; ++i) vals[i] = x[tid + 256 * i];

  __shared__ float red[8];
  const int wid = tid >> 5, lane = tid & 31;

  float m = -3.0e38f;
#pragma unroll
  for (int i = 0; i < 8; ++i) m = fmaxf(m, vals[i]);
#pragma unroll
  for (int off = 16; off > 0; off >>= 1) m = fmaxf(m, __shfl_xor(m, off, 32));
  if (lane == 0) red[wid] = m;
  __syncthreads();
  m = red[0];
#pragma unroll
  for (int j = 1; j < 8; ++j) m = fmaxf(m, red[j]);

  float s = 0.0f;
#pragma unroll
  for (int i = 0; i < 8; ++i) { vals[i] = __expf(vals[i] - m); s += vals[i]; }
#pragma unroll
  for (int off = 16; off > 0; off >>= 1) s += __shfl_xor(s, off, 32);
  __syncthreads();                      // red[] reuse
  if (lane == 0) red[wid] = s;
  __syncthreads();
  s = red[0] + red[1] + red[2] + red[3] + red[4] + red[5] + red[6] + red[7];
  const float inv = 1.0f / s;
#pragma unroll
  for (int i = 0; i < 8; ++i)
    out[tid + 256 * i] = (bf16)(vals[i] * inv + it[tid + 256 * i]);
}

extern "C" void kernel_launch(void* const* d_in, const int* in_sizes, int n_in,
                              void* d_out, int out_size, void* d_ws, size_t ws_size,
                              hipStream_t stream) {
  const int B = 4, S = 2048, D = 1024;
  const float* X         = (const float*)d_in[0];
  const float* intensity = (const float*)d_in[1];
  const float* Wq = (const float*)d_in[2];
  const float* bq = (const float*)d_in[3];
  const float* Wk = (const float*)d_in[4];
  const float* bk = (const float*)d_in[5];
  const float* Wv = (const float*)d_in[6];
  const float* bv = (const float*)d_in[7];
  const float* Wo = (const float*)d_in[8];
  const float* bo = (const float*)d_in[9];
  float* out = (float*)d_out;
  (void)in_sizes; (void)n_in; (void)out_size; (void)ws_size;

  char* ws = (char*)d_ws;
  bf16*  qb      = (bf16*) (ws + (size_t)  0 * (1 << 20));   // [8192 x 1024] bf16, 16 MiB
  bf16*  kb      = (bf16*) (ws + (size_t) 16 * (1 << 20));   // [8192 x 1024] bf16, 16 MiB
  bf16*  vT      = (bf16*) (ws + (size_t) 32 * (1 << 20));   // [1024 x 8192] bf16, 16 MiB
  bf16*  attn    = (bf16*) (ws + (size_t) 48 * (1 << 20));   // [4 x 2048 x 2048] bf16, 32 MiB
  float* scoresF = (float*)(ws + (size_t) 80 * (1 << 20));   // [4 x 2048 x 2048] f32, 64 MiB
  bf16*  av      = (bf16*) (ws + (size_t)144 * (1 << 20));   // [8192 x 1024] bf16, 16 MiB

  const dim3 blk(256);
  const long MS = (long)B * S;          // 8192 rows of X / activations

  // q = X Wq^T + bq ; k = X Wk^T + bk   (f32 in -> bf16 out)
  gemm_wmma_kernel<true, true, false><<<dim3(D / TILE_N, MS / TILE_M, 1), blk, 0, stream>>>(
      X, D, 0, Wq, D, 0, qb, D, 0, bq, 1, 1.0f, D);
  gemm_wmma_kernel<true, true, false><<<dim3(D / TILE_N, MS / TILE_M, 1), blk, 0, stream>>>(
      X, D, 0, Wk, D, 0, kb, D, 0, bk, 1, 1.0f, D);
  // vT[n][m] = sum_k Wv[n][k] X[m][k] + bv[n]  (bias per output ROW)
  gemm_wmma_kernel<true, true, false><<<dim3(MS / TILE_N, D / TILE_M, 1), blk, 0, stream>>>(
      Wv, D, 0, X, D, 0, vT, MS, 0, bv, 2, 1.0f, D);

  // scores[b] = q[b] k[b]^T / 32   (bf16 in -> f32 out), batched over z
  gemm_wmma_kernel<false, false, true><<<dim3(S / TILE_N, S / TILE_M, B), blk, 0, stream>>>(
      qb, D, (long)S * D, kb, D, (long)S * D, scoresF, S, (long)S * S,
      nullptr, 0, 1.0f / 32.0f, D);

  // attn = softmax(scores) + intensity  -> bf16
  softmax_intensity_kernel<<<dim3(S, B), blk, 0, stream>>>(scoresF, intensity, attn);

  // av[b] = attn[b] @ v[b] ; B-operand rows are vT rows (contiguous K per lane)
  gemm_wmma_kernel<false, false, false><<<dim3(D / TILE_N, S / TILE_M, B), blk, 0, stream>>>(
      attn, S, (long)S * S, vT, MS, (long)S, av, D, (long)S * D,
      nullptr, 0, 1.0f, S);

  // out = av Wo^T + bo  (bf16 x f32 -> f32 to d_out)
  gemm_wmma_kernel<false, true, true><<<dim3(D / TILE_N, MS / TILE_M, 1), blk, 0, stream>>>(
      av, D, 0, Wo, D, 0, out, D, 0, bo, 1, 1.0f, D);
}
